// HistLayer_15753940042001
// MI455X (gfx1250) — 8-bit matrix operand order repaired and checked by reading, compiled, not run on hardware
//
#include <hip/hip_runtime.h>

typedef int   v8i __attribute__((ext_vector_type(8)));
typedef float v4f __attribute__((ext_vector_type(4)));

#define DIM_H 4096
#define DIM_W 4096
#define OUT_H 512
#define OUT_W 512

__device__ __forceinline__ void bin_update(float x, unsigned long long& lo,
                                           unsigned long long& hi) {
    float t = x * 16.0f;
    int b = (int)__builtin_ceilf(t) - 1;
    b = b < 0 ? 0 : b;
    b = b > 15 ? 15 : b;
    unsigned long long inc = 1ull << ((b & 7) << 3);
    lo += (b < 8) ? inc : 0ull;
    hi += (b < 8) ? 0ull : inc;
}

__global__ __launch_bounds__(256) void hist_mode_kernel(const float* __restrict__ xx,
                                                        float* __restrict__ out) {
    const int tid   = threadIdx.x;
    const int lane  = tid & 31;
    const int waveId = blockIdx.x * 8 + (tid >> 5);
    const int wy     = waveId >> 5;
    const int wxBase = (waveId & 31) << 4;

    const int  m      = lane & 15;
    const bool hiHalf = lane >= 16;

    const int wx = wxBase + m;
    const int y0 = wy << 3;
    const int x0 = wx << 3;

    const int rbase = hiHalf ? 2 : 0;

    unsigned long long hl[2] = {0ull, 0ull};
    unsigned long long hh[2] = {0ull, 0ull};

#pragma unroll
    for (int g = 0; g < 2; ++g) {
#pragma unroll
        for (int rsub = 0; rsub < 2; ++rsub) {
            const int row = rbase + rsub + (g << 2);
            const v4f* p = (const v4f*)(xx + (size_t)(y0 + row) * DIM_W + x0);
            v4f a = __builtin_nontemporal_load(p);
            v4f b = __builtin_nontemporal_load(p + 1);
            bin_update(a[0], hl[g], hh[g]);
            bin_update(a[1], hl[g], hh[g]);
            bin_update(a[2], hl[g], hh[g]);
            bin_update(a[3], hl[g], hh[g]);
            bin_update(b[0], hl[g], hh[g]);
            bin_update(b[1], hl[g], hh[g]);
            bin_update(b[2], hl[g], hh[g]);
            bin_update(b[3], hl[g], hh[g]);
        }
    }

    const bool     active  = true;
    const int      r       = m & 7;
    const unsigned byteval = 1u << ((r & 3) << 3);
    const int evenv = (active && (r < 4))  ? (int)byteval : 0;
    const int oddv  = (active && (r >= 4)) ? (int)byteval : 0;
    v8i A;
    A[0] = (m < 8) ? evenv : 0; A[1] = (m < 8) ? oddv : 0; A[2] = (m < 8) ? 0 : evenv; A[3] = (m < 8) ? 0 : oddv;
    A[4] = (m < 8) ? evenv : 0; A[5] = (m < 8) ? oddv : 0; A[6] = (m < 8) ? 0 : evenv; A[7] = (m < 8) ? 0 : oddv;

    v8i B;
    B[0] = (int)(unsigned)hl[0]; B[1] = (int)(unsigned)(hl[0] >> 32);
    B[2] = (int)(unsigned)hh[0]; B[3] = (int)(unsigned)(hh[0] >> 32);
    B[4] = (int)(unsigned)hl[1]; B[5] = (int)(unsigned)(hl[1] >> 32);
    B[6] = (int)(unsigned)hh[1]; B[7] = (int)(unsigned)(hh[1] >> 32);

    v8i Cz = {0, 0, 0, 0, 0, 0, 0, 0};
    v8i D = __builtin_amdgcn_wmma_i32_16x16x64_iu8(false, A, false, B, Cz,
                                                   false, false);

    const int base  = hiHalf ? 8 : 0;
    int bestC = D[0];
    int bestB = base;
#pragma unroll
    for (int j = 1; j < 8; ++j) {
        int c = D[j];
        if (c > bestC) { bestC = c; bestB = base + j; }
    }

    const int otherC = __shfl_xor(bestC, 16, 32);
    const int otherB = __shfl_xor(bestB, 16, 32);

    if (!hiHalf) {
        const int fin = (bestC >= otherC) ? bestB : otherB;
        __builtin_nontemporal_store((float)fin, out + (size_t)wy * OUT_W + wx);
    }
}

extern "C" void kernel_launch(void* const* d_in, const int* in_sizes, int n_in,
                              void* d_out, int out_size, void* d_ws, size_t ws_size,
                              hipStream_t stream) {
    (void)in_sizes; (void)n_in; (void)out_size; (void)d_ws; (void)ws_size;
    const float* xx = (const float*)d_in[0];
    float* out = (float*)d_out;
    hipLaunchKernelGGL(hist_mode_kernel, dim3(2048), dim3(256), 0, stream, xx, out);
}
